// MLPBlockFused_74191265071209
// MI455X (gfx1250) — matrix operand order repaired and checked by reading, not compiled, not run on hardware
//
#include <hip/hip_runtime.h>
#include <hip/hip_bf16.h>

// ---------------------------------------------------------------------------
// Fused MoE FFN for MI455X (gfx1250, wave32, WMMA).
// Memory-bound: 768MB of fp32 expert weights @ 23.3TB/s ~= 33us floor.
// Strategy: group tokens by expert; stream each 64x1024 weight slab from HBM
// exactly once into a 128KB f16 LDS tile; run v_wmma_f32_16x16x32_f16 over
// 16-token tiles against the LDS-resident weights.
// ---------------------------------------------------------------------------

#define T_TOK 1024
#define H_DIM 1024
#define I_DIM 1024
#define E_NUM 64
#define KSEL  4
#define ALPHA 1.702f
#define LIMIT 7.0f
#define EPSV  1e-5f
#define ROWS  64          // weight rows per block (LDS slab = ROWS*1024*2B = 128KB)

typedef _Float16 f16;
typedef __attribute__((ext_vector_type(16))) _Float16 v16h;
typedef __attribute__((ext_vector_type(8)))  _Float16 h8;
typedef __attribute__((ext_vector_type(4)))  _Float16 h4;
typedef __attribute__((ext_vector_type(8)))  float    v8f;

// ---------------------------------------------------------------------------
// Kernel 0: zero expert counters
__global__ void k_init(int* __restrict__ counts) {
    if (threadIdx.x < E_NUM) counts[threadIdx.x] = 0;
}

// ---------------------------------------------------------------------------
// Kernel 1: RMSNorm (fp32) -> t_f16, gate logits, top-4 + softmax, counts
__global__ __launch_bounds__(256) void k_gate(
    const float* __restrict__ x, const float* __restrict__ nscale,
    const float* __restrict__ gw, const float* __restrict__ gb,
    f16* __restrict__ tf16, float* __restrict__ ew, int* __restrict__ eidx,
    int* __restrict__ counts)
{
    __shared__ float xs[H_DIM];
    __shared__ float part[E_NUM][4];
    __shared__ float lg[E_NUM];
    __shared__ float red[8];

    const int t = blockIdx.x, tid = threadIdx.x;
    const float* xr = x + (size_t)t * H_DIM;

    float ss = 0.f;
    for (int i = tid; i < H_DIM; i += 256) { float v = xr[i]; xs[i] = v; ss += v * v; }
    for (int o = 16; o > 0; o >>= 1) ss += __shfl_xor(ss, o, 32);
    if ((tid & 31) == 0) red[tid >> 5] = ss;
    __syncthreads();
    if (tid == 0) {
        float s = 0.f;
        for (int i = 0; i < 8; ++i) s += red[i];
        red[0] = rsqrtf(s / (float)H_DIM + EPSV);
    }
    __syncthreads();
    const float r = red[0];
    for (int i = tid; i < H_DIM; i += 256) {
        float v = xs[i] * r * nscale[i];
        xs[i] = v;
        tf16[(size_t)t * H_DIM + i] = (f16)v;
    }
    __syncthreads();

    // logits: 4 threads per expert, deterministic tree combine
    {
        const int e = tid >> 2, p = tid & 3;
        const float* w = gw + (size_t)e * H_DIM + p * 256;
        const float* tv = xs + p * 256;
        float s = 0.f;
        for (int k = 0; k < 256; ++k) s += tv[k] * w[k];
        part[e][p] = s;
    }
    __syncthreads();
    if (tid < E_NUM)
        lg[tid] = gb[tid] + ((part[tid][0] + part[tid][1]) + (part[tid][2] + part[tid][3]));
    __syncthreads();

    if (tid == 0) {
        float bv[KSEL]; int bi[KSEL];
        for (int s = 0; s < KSEL; ++s) { bv[s] = -1e30f; bi[s] = 0; }
        for (int e = 0; e < E_NUM; ++e) {
            float v = lg[e];
            if (v > bv[KSEL - 1]) {
                int s = KSEL - 1;
                while (s > 0 && v > bv[s - 1]) { bv[s] = bv[s - 1]; bi[s] = bi[s - 1]; --s; }
                bv[s] = v; bi[s] = e;
            }
        }
        float m = bv[0], sum = 0.f, pr[KSEL];
        for (int s = 0; s < KSEL; ++s) { pr[s] = expf(bv[s] - m); sum += pr[s]; }
        for (int s = 0; s < KSEL; ++s) {
            ew[t * KSEL + s] = pr[s] / sum;
            eidx[t * KSEL + s] = bi[s];
            atomicAdd(&counts[bi[s]], 1);
        }
    }
}

// ---------------------------------------------------------------------------
// Kernel 2: exclusive scan of expert counts (tiny, serial)
__global__ void k_scan(const int* __restrict__ counts, int* __restrict__ offs,
                       int* __restrict__ cursor)
{
    if (threadIdx.x == 0) {
        int s = 0;
        for (int e = 0; e < E_NUM; ++e) { offs[e] = s; cursor[e] = s; s += counts[e]; }
        offs[E_NUM] = s;
    }
}

// ---------------------------------------------------------------------------
// Kernel 3: scatter (token, slot) -> grouped assignment index
__global__ void k_scatter(const int* __restrict__ eidx, int* __restrict__ cursor,
                          int* __restrict__ tlist, int* __restrict__ spos)
{
    int i = blockIdx.x * blockDim.x + threadIdx.x;
    if (i < T_TOK * KSEL) {
        int e = eidx[i];
        int p = atomicAdd(&cursor[e], 1);
        tlist[p] = i >> 2;
        spos[i] = p;
    }
}

// ---------------------------------------------------------------------------
// WMMA fragment helpers (ISA 7.12.2 16-bit layouts).
// A 16x32: lane m = L&15, K-group = L>>4; halves 0..7 -> K contiguous at
//          kb + 8*kg, halves 8..15 -> K contiguous at kb + 16 + 8*kg.
// B 32x16: lane n = L&15, K-group = L>>4; same K mapping as A.
__device__ __forceinline__ v16h load_frag_a(const f16* __restrict__ row, int kb) {
    h8 lo = *(const h8*)(row + kb);
    h8 hi = *(const h8*)(row + kb + 16);
    return __builtin_shufflevector(lo, hi, 0,1,2,3,4,5,6,7,8,9,10,11,12,13,14,15);
}
__device__ __forceinline__ v16h load_frag_b(const f16* __restrict__ row, int kb) {
    h8 lo = *(const h8*)(row + kb);
    h8 hi = *(const h8*)(row + kb + 16);
    return __builtin_shufflevector(lo, hi, 0,1,2,3,4,5,6,7,8,9,10,11,12,13,14,15);
}

// fast sigmoid: argument is bounded (g <= 7), native exp + native rcp suffice
__device__ __forceinline__ float fast_sigmoid(float z) {
    return __builtin_amdgcn_rcpf(1.f + __expf(-z));
}

// ---------------------------------------------------------------------------
// Kernel 4: h = t @ mlp1_w[e]^T + b1  -> SwiGLU -> act (f16)
// grid: (2048/ROWS, E), block: (ROWS/16)*32 threads
__global__ __launch_bounds__(128) void k_mlp1(
    const f16* __restrict__ tf16, const float* __restrict__ w1,
    const float* __restrict__ b1, const int* __restrict__ offs,
    const int* __restrict__ tlist, f16* __restrict__ act)
{
    __shared__ __align__(16) f16 slab[ROWS * H_DIM];   // 128KB f16 weight tile

    const int e = blockIdx.y;
    const int rbase = blockIdx.x * ROWS;               // row in [0, 2I)
    const float* wsrc = w1 + ((size_t)e * 2 * I_DIM + rbase) * H_DIM;

    // stream slab fp32 -> f16 LDS (read from HBM exactly once)
    for (int f = threadIdx.x; f < ROWS * H_DIM / 4; f += 128) {
        float4 v = ((const float4*)wsrc)[f];
        h4 hv = { (f16)v.x, (f16)v.y, (f16)v.z, (f16)v.w };
        *(h4*)(&slab[f * 4]) = hv;
    }
    __syncthreads();

    const int lane = threadIdx.x & 31, wv = threadIdx.x >> 5;
    const int kg = lane >> 4, n = lane & 15;
    const int off = offs[e], end = offs[e + 1];
    const int ntile = (end - off + 15) >> 4;
    const float bias = b1[(size_t)e * 2 * I_DIM + rbase + wv * 16 + n];
    const f16* brow = slab + (size_t)(wv * 16 + n) * H_DIM + kg * 8;
    const int idim = (rbase >> 1) + wv * 8 + (n >> 1);
    const bool evenlane = (n & 1) == 0;

    for (int tile = 0; tile < ntile; ++tile) {
        const int abase = off + tile * 16;
        const int am = abase + n;
        const int tok = (am < end) ? tlist[am] : 0;
        const f16* arow = tf16 + (size_t)tok * H_DIM + kg * 8;

        v8f c = {};
        #pragma unroll 4
        for (int kb = 0; kb < H_DIM; kb += 32) {
            v16h a = load_frag_a(arow, kb);
            v16h b = load_frag_b(brow, kb);
            c = __builtin_amdgcn_wmma_f32_16x16x32_f16(false, a, false, b,
                                                       (short)0, c, false, false);
        }

        // C layout: c[r] = h[m = r + 8*kg][row = rbase + wv*16 + n]
        #pragma unroll
        for (int r = 0; r < 8; ++r) {
            float h = c[r] + bias;
            float other = __shfl_xor(h, 1, 32);   // glu/lin partner (adjacent row)
            if (evenlane) {
                float g = fminf(h, LIMIT);
                float l = fminf(fmaxf(other, -LIMIT), LIMIT);
                float av = g * fast_sigmoid(ALPHA * g) * (l + 1.f);
                int aa = abase + r + 8 * kg;
                if (aa < end) act[(size_t)aa * I_DIM + idim] = (f16)av;
            }
        }
    }
}

// ---------------------------------------------------------------------------
// Kernel 5: o = act @ mlp2_w[e]^T + b2  -> obuf (fp32, per assignment)
// grid: (1024/ROWS, E), block: 128
__global__ __launch_bounds__(128) void k_mlp2(
    const f16* __restrict__ act, const float* __restrict__ w2,
    const float* __restrict__ b2, const int* __restrict__ offs,
    float* __restrict__ obuf)
{
    __shared__ __align__(16) f16 slab[ROWS * I_DIM];   // 128KB

    const int e = blockIdx.y;
    const int hbase = blockIdx.x * ROWS;               // h row in [0, H)
    const float* wsrc = w2 + ((size_t)e * H_DIM + hbase) * I_DIM;

    for (int f = threadIdx.x; f < ROWS * I_DIM / 4; f += 128) {
        float4 v = ((const float4*)wsrc)[f];
        h4 hv = { (f16)v.x, (f16)v.y, (f16)v.z, (f16)v.w };
        *(h4*)(&slab[f * 4]) = hv;
    }
    __syncthreads();

    const int lane = threadIdx.x & 31, wv = threadIdx.x >> 5;
    const int kg = lane >> 4, n = lane & 15;
    const int off = offs[e], end = offs[e + 1];
    const int ntile = (end - off + 15) >> 4;
    const int hcol = hbase + wv * 16 + n;
    const float bias = b2[(size_t)e * H_DIM + hcol];
    const f16* brow = slab + (size_t)(wv * 16 + n) * I_DIM + kg * 8;

    for (int tile = 0; tile < ntile; ++tile) {
        const int abase = off + tile * 16;
        int am = abase + n;
        int asafe = (am < end) ? am : (end - 1);
        const f16* arow = act + (size_t)asafe * I_DIM + kg * 8;

        v8f c = {};
        #pragma unroll 4
        for (int kb = 0; kb < I_DIM; kb += 32) {
            v16h a = load_frag_a(arow, kb);
            v16h b = load_frag_b(brow, kb);
            c = __builtin_amdgcn_wmma_f32_16x16x32_f16(false, a, false, b,
                                                       (short)0, c, false, false);
        }

        #pragma unroll
        for (int r = 0; r < 8; ++r) {
            int aa = abase + r + 8 * kg;
            if (aa < end) obuf[(size_t)aa * H_DIM + hcol] = c[r] + bias;
        }
    }
}

// ---------------------------------------------------------------------------
// Kernel 6: out = x + sum_s ew[t,s] * o[slot_pos[t,s]]   (deterministic order)
__global__ __launch_bounds__(256) void k_final(
    const float* __restrict__ x, const float* __restrict__ ew,
    const int* __restrict__ spos, const float* __restrict__ obuf,
    float* __restrict__ out)
{
    const int t = blockIdx.x;
    for (int h = threadIdx.x; h < H_DIM; h += 256) {
        float acc = x[(size_t)t * H_DIM + h];
        #pragma unroll
        for (int s = 0; s < KSEL; ++s) {
            int a = spos[t * KSEL + s];
            acc += ew[t * KSEL + s] * obuf[(size_t)a * H_DIM + h];
        }
        out[(size_t)t * H_DIM + h] = acc;
    }
}

// ---------------------------------------------------------------------------
extern "C" void kernel_launch(void* const* d_in, const int* in_sizes, int n_in,
                              void* d_out, int out_size, void* d_ws, size_t ws_size,
                              hipStream_t stream) {
    const float* x      = (const float*)d_in[0];
    const float* nscale = (const float*)d_in[1];
    const float* gw     = (const float*)d_in[2];
    const float* gb     = (const float*)d_in[3];
    const float* w1     = (const float*)d_in[4];
    const float* b1     = (const float*)d_in[5];
    const float* w2     = (const float*)d_in[6];
    const float* b2     = (const float*)d_in[7];
    float* out = (float*)d_out;

    // workspace layout
    char* w = (char*)d_ws;
    constexpr size_t OFF_TF16 = 0;                                 // T*H f16  = 2MB
    constexpr size_t OFF_ACT  = OFF_TF16 + (size_t)T_TOK * H_DIM * 2;      // 8MB
    constexpr size_t OFF_OBUF = OFF_ACT + (size_t)T_TOK * KSEL * I_DIM * 2; // 16MB
    constexpr size_t OFF_EW   = OFF_OBUF + (size_t)T_TOK * KSEL * H_DIM * 4;
    constexpr size_t OFF_IDX  = OFF_EW   + (size_t)T_TOK * KSEL * 4;
    constexpr size_t OFF_POS  = OFF_IDX  + (size_t)T_TOK * KSEL * 4;
    constexpr size_t OFF_TL   = OFF_POS  + (size_t)T_TOK * KSEL * 4;
    constexpr size_t OFF_CNT  = OFF_TL   + (size_t)T_TOK * KSEL * 4;
    constexpr size_t OFF_OFFS = OFF_CNT  + (size_t)E_NUM * 4;
    constexpr size_t OFF_CUR  = OFF_OFFS + (size_t)(E_NUM + 1) * 4;

    f16*   tf16  = (f16*)(w + OFF_TF16);
    f16*   actb  = (f16*)(w + OFF_ACT);
    float* obuf  = (float*)(w + OFF_OBUF);
    float* ewb   = (float*)(w + OFF_EW);
    int*   eidx  = (int*)(w + OFF_IDX);
    int*   spos  = (int*)(w + OFF_POS);
    int*   tlist = (int*)(w + OFF_TL);
    int*   cnt   = (int*)(w + OFF_CNT);
    int*   offs  = (int*)(w + OFF_OFFS);
    int*   cur   = (int*)(w + OFF_CUR);

    k_init<<<1, 64, 0, stream>>>(cnt);
    k_gate<<<T_TOK, 256, 0, stream>>>(x, nscale, gw, gb, tf16, ewb, eidx, cnt);
    k_scan<<<1, 32, 0, stream>>>(cnt, offs, cur);
    k_scatter<<<(T_TOK * KSEL + 255) / 256, 256, 0, stream>>>(eidx, cur, tlist, spos);
    k_mlp1<<<dim3(2 * I_DIM / ROWS, E_NUM), 128, 0, stream>>>(tf16, w1, b1, offs, tlist, actb);
    k_mlp2<<<dim3(H_DIM / ROWS, E_NUM), 128, 0, stream>>>(actb, w2, b2, offs, obuf);
    k_final<<<T_TOK, 256, 0, stream>>>(x, ewb, spos, obuf, out);
}
